// Transformer_47382079209601
// MI455X (gfx1250) — matrix operand order repaired and checked by reading, not compiled, not run on hardware
//
#include <hip/hip_runtime.h>

// CDNA5 / gfx1250 wave32 WMMA implementation of the transformer block:
//   x = prompt@Wp+bp ; qkv = x@Wqkv+bqkv ; flash-attention (8 heads, d=64) ;
//   out = ctx@Wo+bo
// All GEMMs/attention use v_wmma_f32_16x16x32_f16 (f16 inputs, f32 accum).
// Flash attention double-buffers K tiles with GLOBAL_LOAD_ASYNC_TO_LDS_B128
// (ASYNCcnt) and V tiles with register prefetch, so the per-tile L2 latency
// is hidden behind the WMMA work.

typedef __attribute__((ext_vector_type(16))) _Float16 v16h;
typedef __attribute__((ext_vector_type(8)))  _Float16 h8;
typedef __attribute__((ext_vector_type(4)))  _Float16 h4;
typedef __attribute__((ext_vector_type(8)))  float    v8f;
typedef __attribute__((ext_vector_type(4)))  float    f4;

#define LDSTR 80   // LDS row stride in halves (160B: 16B-aligned rows, de-conflicted)

#define WMMA(a,b,c) __builtin_amdgcn_wmma_f32_16x16x32_f16(false,(a),false,(b),(short)0,(c),false,false)

static __device__ __forceinline__ v16h mkv16(h8 a, h8 b) {
  return __builtin_shufflevector(a, b, 0,1,2,3,4,5,6,7,8,9,10,11,12,13,14,15);
}
// A-matrix fragment (16x32 f16). Lane L holds row M=L%16; halves e map to
// K = e + (e>=8 ? 8 : 0) + 8*(L/16)  (ISA 7.12.2 16-bit A layout).
static __device__ __forceinline__ v16h afrag(const _Float16* rowp, int kk, int hi) {
  const int b0 = kk * 32 + 8 * hi;
  h8 x = *(const h8*)(rowp + b0);
  h8 y = *(const h8*)(rowp + b0 + 16);
  return mkv16(x, y);
}
// B-matrix fragment (32x16 f16). Lane L holds col N=L%16; halves e map to
// K = e + (e>=8 ? 8 : 0) + 8*(L/16). colp points at a [N][K] (transposed-in-LDS) row.
static __device__ __forceinline__ v16h bfrag(const _Float16* colp, int kk, int hi) {
  const _Float16* p = colp + kk * 32 + 8 * hi;
  h8 x = *(const h8*)p;
  h8 y = *(const h8*)(p + 16);
  return mkv16(x, y);
}
static __device__ __forceinline__ v8f vzero8() {
  v8f z = {0.f, 0.f, 0.f, 0.f, 0.f, 0.f, 0.f, 0.f};
  return z;
}
// Async 16B global -> LDS copy (per lane), tracked by ASYNCcnt.
// Generic pointers to __shared__ carry the LDS byte offset in their low 32
// bits (flat-aperture truncation), which is exactly what VDST wants.
static __device__ __forceinline__ void async_ld_b128(const _Float16* gp, _Float16* lp) {
  unsigned int loff = (unsigned int)(unsigned long long)lp;
  asm volatile("global_load_async_to_lds_b128 %0, %1, off"
               :: "v"(loff), "v"(gp) : "memory");
}
static __device__ __forceinline__ void wait_asynccnt0() {
  asm volatile("s_wait_asynccnt 0" ::: "memory");
}

// ---------------------------------------------------------------------------
// Tiled GEMM: C[M,N] = A[M,K] * B[K,N] + bias[N]
// Block tile 128x128, K-step 64, 256 threads = 8 waves (4 M x 2 N),
// wave tile 32x64 = 2x4 WMMA accumulators.
// mode 0: A=fp32(prompt)          -> out16 = X (f16 row-major [M,N])
// mode 1: A=f16(X)                -> scatter Q/K/V f16 [b,h,t,64], Q *= 1/8
// mode 2: A=f16(ctx)              -> out32 fp32 row-major [M,N]
// ---------------------------------------------------------------------------
__global__ __launch_bounds__(256)
void gemm_wmma(const float* __restrict__ A32, const _Float16* __restrict__ A16,
               const float* __restrict__ Bg,  const float* __restrict__ bias,
               float* __restrict__ out32, _Float16* __restrict__ out16,
               _Float16* __restrict__ qout, _Float16* __restrict__ kout,
               _Float16* __restrict__ vout,
               int M, int N, int K, int mode)
{
  __shared__ __align__(16) _Float16 sA[128 * LDSTR]; // [row][k]
  __shared__ __align__(16) _Float16 sB[128 * LDSTR]; // [n][k] (transposed)

  const int tid  = threadIdx.x;
  const int w    = tid >> 5, lane = tid & 31;
  const int lo   = lane & 15, hi = lane >> 4;
  const int wm   = w & 3, wn = w >> 2;
  const int m0   = blockIdx.y * 128;
  const int n0   = blockIdx.x * 128;

  v8f acc[2][4];
#pragma unroll
  for (int mi = 0; mi < 2; ++mi)
#pragma unroll
    for (int nj = 0; nj < 4; ++nj) acc[mi][nj] = vzero8();

  for (int kt = 0; kt < K; kt += 64) {
    __syncthreads();
    if (mode == 0) {                     // fp32 A -> convert to f16
      const float* Ab = A32 + (size_t)m0 * K + kt;
#pragma unroll
      for (int it = 0; it < 8; ++it) {   // 2048 float4 chunks of 128x64
        int c = it * 256 + tid;
        int row = c >> 4, col = (c & 15) * 4;
        f4 v = *(const f4*)(Ab + (size_t)row * K + col);
        h4 hv;
#pragma unroll
        for (int i = 0; i < 4; ++i) hv[i] = (_Float16)v[i];
        *(h4*)(sA + row * LDSTR + col) = hv;
      }
    } else {                             // f16 A
      const _Float16* Ab = A16 + (size_t)m0 * K + kt;
#pragma unroll
      for (int it = 0; it < 4; ++it) {   // 1024 h8 chunks of 128x64
        int c = it * 256 + tid;
        int row = c >> 3, col = (c & 7) * 8;
        *(h8*)(sA + row * LDSTR + col) = *(const h8*)(Ab + (size_t)row * K + col);
      }
    }
    {                                    // B fp32 [64,k x 128,n] -> f16, transposed
      const float* Bb = Bg + (size_t)kt * N + n0;
#pragma unroll
      for (int it = 0; it < 8; ++it) {   // 2048 float4 chunks of 64x128
        int c = it * 256 + tid;
        int kr = c >> 5, nc = (c & 31) * 4;
        f4 v = *(const f4*)(Bb + (size_t)kr * N + nc);
#pragma unroll
        for (int i = 0; i < 4; ++i) sB[(nc + i) * LDSTR + kr] = (_Float16)v[i];
      }
    }
    __syncthreads();
#pragma unroll
    for (int kk = 0; kk < 2; ++kk) {
      v16h af[2], bf[4];
#pragma unroll
      for (int mi = 0; mi < 2; ++mi)
        af[mi] = afrag(sA + (wm * 32 + mi * 16 + lo) * LDSTR, kk, hi);
#pragma unroll
      for (int nj = 0; nj < 4; ++nj)
        bf[nj] = bfrag(sB + (wn * 64 + nj * 16 + lo) * LDSTR, kk, hi);
#pragma unroll
      for (int mi = 0; mi < 2; ++mi)
#pragma unroll
        for (int nj = 0; nj < 4; ++nj)
          acc[mi][nj] = WMMA(af[mi], bf[nj], acc[mi][nj]);
    }
  }

  // Epilogue. C layout: lane holds col N=lo; VGPR r holds row r + 8*hi.
#pragma unroll
  for (int nj = 0; nj < 4; ++nj) {
    const int col = n0 + wn * 64 + nj * 16 + lo;
    const float bv = bias[col];
#pragma unroll
    for (int mi = 0; mi < 2; ++mi) {
      const int rowb = m0 + wm * 32 + mi * 16 + 8 * hi;
#pragma unroll
      for (int r = 0; r < 8; ++r) {
        const int row = rowb + r;
        const float val = acc[mi][nj][r] + bv;
        if (mode == 0) {
          out16[(size_t)row * N + col] = (_Float16)val;
        } else if (mode == 2) {
          out32[(size_t)row * N + col] = val;
        } else {
          const int region = col >> 9;          // 0:Q 1:K 2:V
          const int c2 = col & 511;
          const int hh = c2 >> 6, d = c2 & 63;
          const int b = row >> 12, t = row & 4095;
          _Float16* dst = (region == 0) ? qout : ((region == 1) ? kout : vout);
          const float sc = (region == 0) ? 0.125f : 1.0f;  // 1/sqrt(64) folded into Q
          dst[((((size_t)b * 8 + hh) * 4096 + t) << 6) + d] = (_Float16)(val * sc);
        }
      }
    }
  }
}

// ---------------------------------------------------------------------------
// Flash attention. Q,K,V f16 [b,h,4096,64] (Q pre-scaled by 1/8).
// Grid (T/128, B*H); 256 threads = 8 waves; 16 query rows per wave.
// Double-buffered 64-key tiles: next K tile via async global->LDS copies
// (ASYNCcnt), next V tile via register prefetch + transposed LDS stores.
// Per tile: S = Q K^T (8 WMMA), fp32 online softmax in registers with
// cross-lane butterflies, P through per-wave LDS scratch (C->A layout),
// O += P V (8 WMMA).  Output ctx f16 [b,t, h*64+d] row-major [8192,512].
// ---------------------------------------------------------------------------
__global__ __launch_bounds__(256)
void flash_attn_wmma(const _Float16* __restrict__ Qg,
                     const _Float16* __restrict__ Kg,
                     const _Float16* __restrict__ Vg,
                     _Float16* __restrict__ ctx)
{
  constexpr int T = 4096, D = 64;
  constexpr int TILE = 64 * LDSTR;
  __shared__ __align__(16) _Float16 sK [2 * TILE];        // [buf][key][d]
  __shared__ __align__(16) _Float16 sVT[2 * TILE];        // [buf][d][key]
  __shared__ __align__(16) _Float16 sP [8 * 16 * LDSTR];  // per-wave [row][key]

  const int tid = threadIdx.x;
  const int w = tid >> 5, lane = tid & 31;
  const int lo = lane & 15, hi = lane >> 4;
  const int bh = blockIdx.y;                 // b*8 + h
  const int t0 = blockIdx.x * 128;

  const _Float16* Qb = Qg + (size_t)bh * T * D;
  const _Float16* Kb = Kg + (size_t)bh * T * D;
  const _Float16* Vb = Vg + (size_t)bh * T * D;
  _Float16* pW = sP + w * 16 * LDSTR;

  // This thread's two staging chunks within a 64x64 tile.
  int srow[2], scol[2];
#pragma unroll
  for (int it = 0; it < 2; ++it) {
    int c = it * 256 + tid;                  // 512 h8 chunks
    srow[it] = c >> 3;
    scol[it] = (c & 7) * 8;
  }

  // Q A-fragments, straight from global (rows 128B-aligned).
  v16h qa[2];
  {
    const _Float16* qr = Qb + (size_t)(t0 + w * 16 + lo) * D;
#pragma unroll
    for (int kk = 0; kk < 2; ++kk) {
      const int b0 = kk * 32 + 8 * hi;
      h8 x = *(const h8*)(qr + b0);
      h8 y = *(const h8*)(qr + b0 + 16);
      qa[kk] = mkv16(x, y);
    }
  }

  float m[8], l[8];
  v8f o[4];
#pragma unroll
  for (int r = 0; r < 8; ++r) { m[r] = -INFINITY; l[r] = 0.f; }
#pragma unroll
  for (int j = 0; j < 4; ++j) o[j] = vzero8();

  // Prologue: stage tile 0 (K async, V transposed via registers).
#pragma unroll
  for (int it = 0; it < 2; ++it) {
    async_ld_b128(Kb + (size_t)srow[it] * D + scol[it],
                  sK + srow[it] * LDSTR + scol[it]);
    h8 v = *(const h8*)(Vb + (size_t)srow[it] * D + scol[it]);
#pragma unroll
    for (int i = 0; i < 8; ++i) sVT[(scol[it] + i) * LDSTR + srow[it]] = v[i];
  }
  wait_asynccnt0();
  __syncthreads();

  int ib = 0;
  for (int kt = 0; kt < T; kt += 64, ib ^= 1) {
    const _Float16* kc = sK  + ib * TILE;
    const _Float16* vc = sVT + ib * TILE;
    const int nb = ib ^ 1;
    const bool hasnext = (kt + 64) < T;

    // Kick off next tile: K via async copy, V into registers.
    h8 vpre[2];
    if (hasnext) {
#pragma unroll
      for (int it = 0; it < 2; ++it) {
        async_ld_b128(Kb + (size_t)(kt + 64 + srow[it]) * D + scol[it],
                      sK + nb * TILE + srow[it] * LDSTR + scol[it]);
        vpre[it] = *(const h8*)(Vb + (size_t)(kt + 64 + srow[it]) * D + scol[it]);
      }
    }

    // S = Q K^T for this 16x64 stripe.
    v8f s[4];
#pragma unroll
    for (int j2 = 0; j2 < 4; ++j2) {
      s[j2] = vzero8();
#pragma unroll
      for (int kk = 0; kk < 2; ++kk) {
        v16h kb = bfrag(kc + (j2 * 16 + lo) * LDSTR, kk, hi);
        s[j2] = WMMA(qa[kk], kb, s[j2]);
      }
    }

    // Online softmax (rows r+8*hi live in this lane's half; reduce over 16 lanes).
    float tmax[8];
#pragma unroll
    for (int r = 0; r < 8; ++r)
      tmax[r] = fmaxf(fmaxf(s[0][r], s[1][r]), fmaxf(s[2][r], s[3][r]));
#pragma unroll
    for (int off = 1; off < 16; off <<= 1)
#pragma unroll
      for (int r = 0; r < 8; ++r)
        tmax[r] = fmaxf(tmax[r], __shfl_xor(tmax[r], off));

    float sc[8];
#pragma unroll
    for (int r = 0; r < 8; ++r) {
      float mn = fmaxf(m[r], tmax[r]);
      sc[r] = __expf(m[r] - mn);
      m[r] = mn;
    }
    float rsum[8];
#pragma unroll
    for (int r = 0; r < 8; ++r) rsum[r] = 0.f;
#pragma unroll
    for (int j2 = 0; j2 < 4; ++j2)
#pragma unroll
      for (int r = 0; r < 8; ++r) {
        float p = __expf(s[j2][r] - m[r]);
        s[j2][r] = p;
        rsum[r] += p;
      }
#pragma unroll
    for (int off = 1; off < 16; off <<= 1)
#pragma unroll
      for (int r = 0; r < 8; ++r) rsum[r] += __shfl_xor(rsum[r], off);
#pragma unroll
    for (int r = 0; r < 8; ++r) l[r] = l[r] * sc[r] + rsum[r];
#pragma unroll
    for (int j = 0; j < 4; ++j)
#pragma unroll
      for (int r = 0; r < 8; ++r) o[j][r] *= sc[r];

    // P (C layout) -> per-wave LDS scratch -> A layout fragments.
#pragma unroll
    for (int j2 = 0; j2 < 4; ++j2)
#pragma unroll
      for (int r = 0; r < 8; ++r)
        pW[(r + 8 * hi) * LDSTR + j2 * 16 + lo] = (_Float16)s[j2][r];
    // LDS ops are in-order per wave; fence keeps the compiler from reordering
    // and drains DScnt before the dependent A-fragment loads.
    asm volatile("s_wait_dscnt 0" ::: "memory");

    v16h pa[2];
#pragma unroll
    for (int kk = 0; kk < 2; ++kk)
      pa[kk] = afrag(pW + lo * LDSTR, kk, hi);
#pragma unroll
    for (int j = 0; j < 4; ++j)
#pragma unroll
      for (int kk = 0; kk < 2; ++kk) {
        v16h vb = bfrag(vc + (j * 16 + lo) * LDSTR, kk, hi);
        o[j] = WMMA(pa[kk], vb, o[j]);
      }

    // Complete next-tile staging: transposed V stores into the other buffer.
    if (hasnext) {
#pragma unroll
      for (int it = 0; it < 2; ++it)
#pragma unroll
        for (int i = 0; i < 8; ++i)
          sVT[nb * TILE + (scol[it] + i) * LDSTR + srow[it]] = vpre[it][i];
    }
    wait_asynccnt0();   // K async copies for the next tile have landed
    __syncthreads();    // V stores visible; all waves done with old buffers
  }

  // Normalize and write ctx [b, t, h*64 + d] (f16, row length 512).
  const int bidx = bh >> 3, h = bh & 7;
  float inv[8];
#pragma unroll
  for (int r = 0; r < 8; ++r) inv[r] = 1.0f / l[r];
#pragma unroll
  for (int j = 0; j < 4; ++j)
#pragma unroll
    for (int r = 0; r < 8; ++r) {
      const int t = t0 + w * 16 + r + 8 * hi;
      ctx[(size_t)(bidx * T + t) * 512 + h * 64 + j * 16 + lo] =
          (_Float16)(o[j][r] * inv[r]);
    }
}

// ---------------------------------------------------------------------------
extern "C" void kernel_launch(void* const* d_in, const int* in_sizes, int n_in,
                              void* d_out, int out_size, void* d_ws, size_t ws_size,
                              hipStream_t stream) {
  (void)in_sizes; (void)n_in; (void)out_size; (void)ws_size;
  const float* prompt = (const float*)d_in[0];  // [2,4096,256]
  const float* Wp     = (const float*)d_in[1];  // [256,512]
  const float* bp     = (const float*)d_in[2];  // [512]
  const float* Wqkv   = (const float*)d_in[3];  // [512,1536]
  const float* bqkv   = (const float*)d_in[4];  // [1536]
  const float* Wo     = (const float*)d_in[5];  // [512,256]
  const float* bo     = (const float*)d_in[6];  // [256]
  float* out = (float*)d_out;                   // [2,4096,256] fp32

  // Workspace layout (f16), 40 MB total.
  _Float16* X  = (_Float16*)d_ws;               // [8192,512]
  _Float16* Qh = X  + (size_t)8192 * 512;       // [2,8,4096,64]
  _Float16* Kh = Qh + (size_t)2 * 8 * 4096 * 64;
  _Float16* Vh = Kh + (size_t)2 * 8 * 4096 * 64;
  _Float16* CT = Vh + (size_t)2 * 8 * 4096 * 64; // [8192,512]

  const dim3 blk(256);

  // 1) X = prompt @ Wp + bp          (M=8192, N=512,  K=256)
  gemm_wmma<<<dim3(512 / 128, 8192 / 128), blk, 0, stream>>>(
      prompt, nullptr, Wp, bp, nullptr, X, nullptr, nullptr, nullptr,
      8192, 512, 256, 0);

  // 2) QKV = X @ Wqkv + bqkv -> scatter to Q/K/V heads (Q scaled by 1/8)
  gemm_wmma<<<dim3(1536 / 128, 8192 / 128), blk, 0, stream>>>(
      nullptr, X, Wqkv, bqkv, nullptr, nullptr, Qh, Kh, Vh,
      8192, 1536, 512, 1);

  // 3) Flash attention -> ctx
  flash_attn_wmma<<<dim3(4096 / 128, 16), blk, 0, stream>>>(Qh, Kh, Vh, CT);

  // 4) out = ctx @ Wo + bo           (M=8192, N=256,  K=512)
  gemm_wmma<<<dim3(256 / 128, 8192 / 128), blk, 0, stream>>>(
      nullptr, CT, Wo, bo, out, nullptr, nullptr, nullptr, nullptr,
      8192, 256, 512, 2);
}
